// LatentDenoisingLayer_50560355008869
// MI455X (gfx1250) — matrix operand order repaired and checked by reading, not compiled, not run on hardware
//
#include <hip/hip_runtime.h>
#include <hip/hip_bf16.h>
#include <math.h>

#define BB    4
#define NPB   1536
#define NTOT  (BB*NPB)          // 6144
#define KNN   30
#define INVK  10
#define KK2   30
#define EPN1  40
#define EPN2  30
#define E1    (NTOT*EPN1)       // 245760
#define E2    (NTOT*EPN2)       // 184320
#define HH    8
#define HCC   32
#define AIND  98
#define SORTN 2048
#define NEG_BIG (-1e9f)

typedef __attribute__((ext_vector_type(16))) _Float16 v16h;
typedef __attribute__((ext_vector_type(8)))  float    v8f;

// ---------------- helpers ----------------

__device__ __forceinline__ v8f v8f_zero() {
  v8f z;
#pragma unroll
  for (int q = 0; q < 8; q++) z[q] = 0.f;
  return z;
}

__device__ __forceinline__ float siluf(float x) { return x / (1.f + __expf(-x)); }

// CDNA5 WMMA f16 16x16x32 per-lane fragment index maps (ISA 7.12.2)
// A (16x32, MxK): lanes 0-15 -> M=lane, K in {0..7,16..23}; lanes 16-31 -> M=lane-16, K in {8..15,24..31}
__device__ __forceinline__ int wmmaAK(int lane, int i) {
  int v = i >> 1;
  int base = (v < 4) ? (v << 1) : (16 + ((v - 4) << 1));
  return base + ((lane >> 4) << 3) + (i & 1);
}
// B (32x16, KxN): lanes 0-15 -> N=lane, K in {0..7,16..23}; lanes 16-31 -> N=lane-16, K in {8..15,24..31}
__device__ __forceinline__ int wmmaBK(int lane, int i) {
  return ((lane >> 4) << 3) + ((i < 8) ? i : (8 + i));
}
// C/D (16x16 f32): element r -> M = r + (lane>>4)*8, N = lane&15

__device__ __forceinline__ void atomicMaxF(float* addr, float val) {
  unsigned int* ai = (unsigned int*)addr;
  unsigned int old = *ai;
  while (true) {
    float f = __uint_as_float(old);
    if (f >= val) break;
    unsigned int assumed = old;
    old = atomicCAS(ai, assumed, __float_as_uint(val));
    if (old == assumed) break;
  }
}

// fused node embedding: rows 0..3, channels 0..32 (channel 32: sel for row 0, X[row-1] for rows 1..3)
__device__ __forceinline__ float fememb(const float* __restrict__ ne, const float* __restrict__ sel,
                                        const float* __restrict__ X, int node, int row, int c) {
  if (c < HCC) return ne[node * 4 * HCC + row * HCC + c];
  return (row == 0) ? sel[node] : X[node * 3 + (row - 1)];
}

// edge rotation matrix (rows: b1, v, b2)
__device__ __forceinline__ void edgeRot(const float* __restrict__ X, int src, int dst, float thr, float* R) {
  float vx = X[dst * 3 + 0] - X[src * 3 + 0];
  float vy = X[dst * 3 + 1] - X[src * 3 + 1];
  float vz = X[dst * 3 + 2] - X[src * 3 + 2];
  float d = sqrtf(vx * vx + vy * vy + vz * vz + 1e-12f);
  bool mask = (d == d) && (d < INFINITY) && (d > thr);
  float v0, v1, v2;
  if (mask) { v0 = vx / d; v1 = vy / d; v2 = vz / d; } else { v0 = 0.f; v1 = 1.f; v2 = 0.f; }
  float r0, r1, r2;
  if (fabsf(v0) > 0.99f) { r0 = 0.f; r1 = 0.f; r2 = 1.f; } else { r0 = 1.f; r1 = 0.f; r2 = 0.f; }
  float b0 = v1 * r2 - v2 * r1;
  float b1 = v2 * r0 - v0 * r2;
  float b2 = v0 * r1 - v1 * r0;
  float nb = sqrtf(b0 * b0 + b1 * b1 + b2 * b2 + 1e-12f);
  b0 /= nb; b1 /= nb; b2 /= nb;
  float c0 = v1 * b2 - v2 * b1;
  float c1 = v2 * b0 - v0 * b2;
  float c2 = v0 * b1 - v1 * b0;
  R[0] = b0; R[1] = b1; R[2] = b2;
  R[3] = v0; R[4] = v1; R[5] = v2;
  R[6] = c0; R[7] = c1; R[8] = c2;
}

// ---------------- graph construction ----------------
// MODE 0: graph1 (x_mask masking, KNN + Gumbel inverse-cubic samples, 40 edges/node)
// MODE 1: graph2 (self masking, K2=30 nearest)
template <int MODE>
__global__ void k_graph(const float* __restrict__ X, const unsigned char* __restrict__ xmask,
                        const float* __restrict__ gum, int* __restrict__ sinks) {
  __shared__ unsigned long long keys[SORTN];
  __shared__ float pert[NPB - KNN];
  __shared__ float rv[256];
  __shared__ int   ri[256];
  int row = blockIdx.x;
  int b = row / NPB, i = row - b * NPB;
  int tid = threadIdx.x;
  float xi0 = X[row * 3 + 0], xi1 = X[row * 3 + 1], xi2 = X[row * 3 + 2];
  bool mi = (MODE == 0) ? (xmask[row] != 0) : false;
  for (int j = tid; j < SORTN; j += 256) {
    unsigned long long key;
    if (j < NPB) {
      int gj = b * NPB + j;
      float dx = xi0 - X[gj * 3 + 0];
      float dy = xi1 - X[gj * 3 + 1];
      float dz = xi2 - X[gj * 3 + 2];
      float d = sqrtf(fmaxf(dx * dx + dy * dy + dz * dz, 1e-12f));
      bool infm = (MODE == 0) ? (mi || (xmask[gj] != 0)) : (j == i);
      if (infm) d = INFINITY;
      key = (((unsigned long long)__float_as_uint(d)) << 32) | (unsigned int)j;
    } else {
      key = (((unsigned long long)0x7F800000u) << 32) | (unsigned int)j;
    }
    keys[j] = key;
  }
  __syncthreads();
  // bitonic sort ascending on packed (dist,index) keys
  for (int k = 2; k <= SORTN; k <<= 1) {
    for (int jj = k >> 1; jj > 0; jj >>= 1) {
      for (int t = tid; t < SORTN; t += 256) {
        int partner = t ^ jj;
        if (partner > t) {
          bool up = ((t & k) == 0);
          unsigned long long a = keys[t], c = keys[partner];
          if ((a > c) == up) { keys[t] = c; keys[partner] = a; }
        }
      }
      __syncthreads();
    }
  }
  const int KOUT = (MODE == 0) ? KNN : KK2;
  const int EPN  = (MODE == 0) ? EPN1 : EPN2;
  if (tid < KOUT) {
    int j = (int)(keys[tid] & 0xFFFFFFFFull);
    sinks[row * EPN + tid] = b * NPB + j;
  }
  if (MODE == 0) {
    for (int s = tid; s < NPB - KNN; s += 256) {
      float ds = __uint_as_float((unsigned int)(keys[KNN + s] >> 32));
      float u = gum[(b * NPB + i) * (NPB - KNN) + s];
      pert[s] = -3.f * logf(fmaxf(ds, 1e-6f)) - logf(-logf(u));
    }
    __syncthreads();
    for (int r = 0; r < INVK; r++) {
      float best = -INFINITY; int bi = -1;
      for (int s = tid; s < NPB - KNN; s += 256) {
        float p = pert[s];
        if (p > best) { best = p; bi = s; }
      }
      rv[tid] = best; ri[tid] = bi;
      __syncthreads();
      for (int off = 128; off > 0; off >>= 1) {
        if (tid < off) {
          if (rv[tid + off] > rv[tid]) { rv[tid] = rv[tid + off]; ri[tid] = ri[tid + off]; }
        }
        __syncthreads();
      }
      if (tid == 0) {
        int s = ri[0];
        int j = (int)(keys[KNN + s] & 0xFFFFFFFFull);
        sinks[row * EPN1 + KNN + r] = b * NPB + j;
        pert[s] = -INFINITY;
      }
      __syncthreads();
    }
  }
}

// ---------------- per-edge distance ----------------
__global__ void k_dist(int E, int EPN, const int* __restrict__ sinks, const float* __restrict__ X,
                       float* __restrict__ db) {
  int e = blockIdx.x * blockDim.x + threadIdx.x;
  if (e >= E) return;
  int dst = sinks[e], src = e / EPN;
  float vx = X[dst * 3 + 0] - X[src * 3 + 0];
  float vy = X[dst * 3 + 1] - X[src * 3 + 1];
  float vz = X[dst * 3 + 2] - X[src * 3 + 2];
  db[e] = sqrtf(vx * vx + vy * vy + vz * vz + 1e-12f);
}

// ---------------- logits: silu(ain@Wa1)@Wa2 via WMMA, + segment max ----------------
__global__ void k_logits(int E, int EPN, const int* __restrict__ sinks,
                         const float* __restrict__ ne, const float* __restrict__ sel,
                         const float* __restrict__ X, const float* __restrict__ db,
                         const float* __restrict__ Wa1, const float* __restrict__ Wa2,
                         float thr, float* __restrict__ logits, float* __restrict__ segmax) {
  __shared__ _Float16 ainst[4][16][128];   // K padded 98 -> 128 (zeros)
  __shared__ _Float16 hst[4][16][40];
  __shared__ _Float16 wa1f[4][2][32][16];  // B-fragments of Wa1, pre-laid-out
  __shared__ _Float16 wa2f[32][16];        // B-fragments of Wa2
  int tid = threadIdx.x;
  int wave = tid >> 5, lane = tid & 31;
  int ebase = (blockIdx.x * 4 + wave) * 16;

  // ---- block-cooperative weight fragment staging (uniform code) ----
  for (int idx = tid; idx < 4 * 2 * 32 * 16; idx += 128) {
    int i = idx & 15, ln = (idx >> 4) & 31, nt = (idx >> 9) & 1, kb = idx >> 10;
    int k = kb * 32 + wmmaBK(ln, i);
    int col = nt * 16 + (ln & 15);
    wa1f[kb][nt][ln][i] = (_Float16)((k < AIND) ? Wa1[k * HCC + col] : 0.f);
  }
  for (int idx = tid; idx < 32 * 16; idx += 128) {
    int i = idx & 15, ln = idx >> 4;
    int k = wmmaBK(ln, i);
    int col = ln & 15;
    wa2f[ln][i] = (_Float16)((col < HH) ? Wa2[k * HH + col] : 0.f);
  }

  // ---- ain staging: per-segment uniform loops (no per-element branching) ----
  {
    // embedding halves: lane -> row m = lane&15, half = lane>>4 (0:src, 1:dst)
    int m = lane & 15;
    int e = ebase + m;
    int src = e / EPN, dst = sinks[e];
    int node = (lane >> 4) ? dst : src;
    int kb0 = (lane >> 4) * 33;
    for (int kk = 0; kk < 32; kk++)
      ainst[wave][m][kb0 + kk] = (_Float16)ne[node * 128 + kk];
    ainst[wave][m][kb0 + 32] = (_Float16)sel[node];
  }
  if (lane < 16) {
    // RBF(16) for edge = lane
    int m = lane;
    float d = db[ebase + m];
    for (int t = 0; t < 16; t++) {
      float mu = (20.f / 15.f) * (float)t;
      float z = (d - mu) * 0.8f;            // 1/sigma = 1/1.25
      ainst[wave][m][66 + t] = (_Float16)__expf(-z * z);
    }
  } else {
    // posemb(16) for edge = lane-16: freqs 10000^(-f/8) as constants
    int m = lane - 16;
    int e = ebase + m;
    int src = e / EPN, dst = sinks[e];
    float dpos = (float)(dst - src);
    const float fr[8] = {1.f, 0.31622776601683794f, 0.1f, 0.031622776601683791f,
                         0.01f, 0.0031622776601683794f, 0.001f, 0.00031622776601683794f};
    for (int f = 0; f < 8; f++) {
      float ang = dpos * fr[f];
      ainst[wave][m][82 + f] = (_Float16)__cosf(ang);
      ainst[wave][m][90 + f] = (_Float16)__sinf(ang);
    }
  }
  // zero K padding 98..127
  for (int idx = lane; idx < 16 * 30; idx += 32) {
    int m = idx / 30, k = AIND + (idx - m * 30);
    ainst[wave][m][k] = (_Float16)0.f;
  }
  __syncthreads();

  // ---- GEMM1: hidden = ain @ Wa1 ----
  v8f acc[2];
  acc[0] = v8f_zero(); acc[1] = v8f_zero();
#pragma unroll
  for (int kb = 0; kb < 4; kb++) {
    v16h a;
#pragma unroll
    for (int i = 0; i < 16; i++)
      a[i] = ainst[wave][lane & 15][kb * 32 + wmmaAK(lane, i)];
#pragma unroll
    for (int nt = 0; nt < 2; nt++) {
      v16h bf;
#pragma unroll
      for (int i = 0; i < 16; i++) bf[i] = wa1f[kb][nt][lane][i];
      acc[nt] = __builtin_amdgcn_wmma_f32_16x16x32_f16(false, a, false, bf, (short)0, acc[nt], false, false);
    }
  }
  // silu -> LDS (A-fragment relayout for second GEMM)
#pragma unroll
  for (int nt = 0; nt < 2; nt++) {
#pragma unroll
    for (int r = 0; r < 8; r++) {
      int m = r + ((lane >> 4) << 3);
      int col = nt * 16 + (lane & 15);
      hst[wave][m][col] = (_Float16)siluf(acc[nt][r]);
    }
  }
  __syncthreads();

  // ---- GEMM2: logits = silu(hidden) @ Wa2 ----
  v16h a2, b2f;
#pragma unroll
  for (int i = 0; i < 16; i++)
    a2[i] = hst[wave][lane & 15][wmmaAK(lane, i)];
#pragma unroll
  for (int i = 0; i < 16; i++) b2f[i] = wa2f[lane][i];
  v8f lg = v8f_zero();
  lg = __builtin_amdgcn_wmma_f32_16x16x32_f16(false, a2, false, b2f, (short)0, lg, false, false);
  int col = lane & 15;
  if (col < HH) {
#pragma unroll
    for (int r = 0; r < 8; r++) {
      int m = r + ((lane >> 4) << 3);
      int e = ebase + m;
      float d = db[e];
      bool mask = (d == d) && (d < INFINITY) && (d > thr);
      float v = mask ? lg[r] : NEG_BIG;
      logits[e * HH + col] = v;
      atomicMaxF(&segmax[sinks[e] * HH + col], v);
    }
  }
}

// ---------------- softmax numerator + segment sum ----------------
__global__ void k_psum(int E, const int* __restrict__ sinks, const float* __restrict__ db, float thr,
                       const float* __restrict__ segmax, float* __restrict__ logits,
                       float* __restrict__ segsum) {
  int t = blockIdx.x * blockDim.x + threadIdx.x;
  if (t >= E * HH) return;
  int e = t >> 3;
  float d = db[e];
  bool mask = (d == d) && (d < INFINITY) && (d > thr);
  int dst = sinks[e];
  float m = segmax[dst * HH + (t & 7)];
  if (m < -1e29f) m = 0.f;
  float p = mask ? __expf(logits[t] - m) : 0.f;
  logits[t] = p;
  if (p != 0.f) atomicAdd(&segsum[dst * HH + (t & 7)], p);
}

// ---------------- values: (rotated xm)@Wv via WMMA, scale by attn, back-rotate, scatter ----------------
__global__ void k_value(int E, int EPN, const int* __restrict__ sinks,
                        const float* __restrict__ ne, const float* __restrict__ sel,
                        const float* __restrict__ X, const float* __restrict__ Wv,
                        const float* __restrict__ pbuf, const float* __restrict__ segsum,
                        float thr, float* __restrict__ msgacc) {
  __shared__ _Float16 xst[4][16][96];      // K padded 66 -> 96 (zeros)
  __shared__ float    mst[4][16][64];
  __shared__ _Float16 wvf[3][4][32][16];   // B-fragments of Wv, pre-laid-out
  int tid = threadIdx.x;
  int wave = tid >> 5, lane = tid & 31;
  int ebase = (blockIdx.x * 4 + wave) * 4;   // 4 edges per wave (16 rows)

  // ---- block-cooperative weight fragment staging ----
  for (int idx = tid; idx < 3 * 4 * 32 * 16; idx += 128) {
    int i = idx & 15, ln = (idx >> 4) & 31, nt = (idx >> 9) & 3, kb = idx >> 11;
    int k = kb * 32 + wmmaBK(ln, i);
    int col = nt * 16 + (ln & 15);
    wvf[kb][nt][ln][i] = (_Float16)((k < 66) ? Wv[k * 64 + col] : 0.f);
  }

  // ---- stage xm (16 rows x 66) into LDS; lane l: row l&15, k-half l>>4 ----
  {
    int m = lane & 15;
    int e = ebase + (m >> 2), r = m & 3;
    int src = e / EPN, dst = sinks[e];
    int node = (lane >> 4) ? dst : src;
    int kbase = (lane >> 4) * 33;
    float R[9];
    if (r > 0) edgeRot(X, src, dst, thr, R);
    for (int kk = 0; kk < 33; kk++) {
      float val;
      if (r == 0) val = fememb(ne, sel, X, node, 0, kk);
      else {
        int a = r - 1;
        val = R[a * 3 + 0] * fememb(ne, sel, X, node, 1, kk)
            + R[a * 3 + 1] * fememb(ne, sel, X, node, 2, kk)
            + R[a * 3 + 2] * fememb(ne, sel, X, node, 3, kk);
      }
      xst[wave][m][kbase + kk] = (_Float16)val;
    }
  }
  // zero K padding 66..95
  for (int idx = lane; idx < 16 * 30; idx += 32) {
    int m = idx / 30, k = 66 + (idx - m * 30);
    xst[wave][m][k] = (_Float16)0.f;
  }
  __syncthreads();

  // ---- GEMM: v = xm @ Wv ----
  v8f acc[4];
#pragma unroll
  for (int q = 0; q < 4; q++) acc[q] = v8f_zero();
#pragma unroll
  for (int kb = 0; kb < 3; kb++) {
    v16h a;
#pragma unroll
    for (int i = 0; i < 16; i++)
      a[i] = xst[wave][lane & 15][kb * 32 + wmmaAK(lane, i)];
#pragma unroll
    for (int nt = 0; nt < 4; nt++) {
      v16h bf;
#pragma unroll
      for (int i = 0; i < 16; i++) bf[i] = wvf[kb][nt][lane][i];
      acc[nt] = __builtin_amdgcn_wmma_f32_16x16x32_f16(false, a, false, bf, (short)0, acc[nt], false, false);
    }
  }
  // scale by attention weight, stage to LDS
#pragma unroll
  for (int nt = 0; nt < 4; nt++) {
    int col = nt * 16 + (lane & 15);
    int h = col >> 3;
#pragma unroll
    for (int r = 0; r < 8; r++) {
      int m = r + ((lane >> 4) << 3);
      int e = ebase + (m >> 2);
      int dst = sinks[e];
      float aw = pbuf[e * HH + h] / (segsum[dst * HH + h] + 1e-9f);
      mst[wave][m][col] = aw * acc[nt][r];
    }
  }
  __syncthreads();
  // back-rotate (R^T) and accumulate per-destination message
  {
    int m = lane & 15;
    int c0 = (lane >> 4) * 32;
    int e = ebase + (m >> 2), r = m & 3;
    int src = e / EPN, dst = sinks[e];
    int mb = m & ~3;
    float R[9];
    if (r > 0) edgeRot(X, src, dst, thr, R);
    for (int cc = 0; cc < 32; cc++) {
      int c = c0 + cc;
      float val;
      if (r == 0) val = mst[wave][m][c];
      else {
        int i = r - 1;
        val = R[0 * 3 + i] * mst[wave][mb + 1][c]
            + R[1 * 3 + i] * mst[wave][mb + 2][c]
            + R[2 * 3 + i] * mst[wave][mb + 3][c];
      }
      atomicAdd(&msgacc[dst * 256 + r * 64 + c], val);
    }
  }
}

// ---------------- output projection + gating ----------------
__global__ void k_outproj(int Cout, const float* __restrict__ msgacc,
                          const float* __restrict__ Wo, const float* __restrict__ Wg,
                          float* __restrict__ outb) {
  int n = blockIdx.x * blockDim.x + threadIdx.x;
  if (n >= NTOT) return;
  float o[4][32];
  for (int r = 0; r < 4; r++)
    for (int c = 0; c < Cout; c++) {
      float s = 0.f;
      for (int k = 0; k < 64; k++) s += msgacc[n * 256 + r * 64 + k] * Wo[k * Cout + c];
      o[r][c] = s;
    }
  for (int c = 0; c < Cout; c++) {
    float g = 0.f;
    for (int k = 0; k < Cout; k++) g += o[0][k] * Wg[k * Cout + c];
    g = siluf(g);
    for (int r = 0; r < 4; r++) outb[n * 4 * Cout + r * Cout + c] = o[r][c] * (1.f + g);
  }
}

__global__ void k_newx(const float* __restrict__ X, const float* __restrict__ sel,
                       const float* __restrict__ outb, float* __restrict__ newX) {
  int n = blockIdx.x * blockDim.x + threadIdx.x;
  if (n >= NTOT) return;
  bool noised = sel[n] > 0.5f;
  for (int k = 0; k < 3; k++)
    newX[n * 3 + k] = noised ? outb[n * 4 + (1 + k)] : X[n * 3 + k];
}

__global__ void k_final(const float* __restrict__ newX, const float* __restrict__ ne,
                        const float* __restrict__ sel, const float* __restrict__ outb,
                        float* __restrict__ dout) {
  int n = blockIdx.x * blockDim.x + threadIdx.x;
  if (n >= NTOT) return;
  for (int k = 0; k < 3; k++) dout[n * 3 + k] = newX[n * 3 + k];
  bool noised = sel[n] > 0.5f;
  float* eo = dout + NTOT * 3;
  for (int t = 0; t < 128; t++)
    eo[n * 128 + t] = noised ? outb[n * 128 + t] : ne[n * 128 + t];
}

__global__ void k_sel_scatter(const int* __restrict__ idx, float* __restrict__ sel, int cnt) {
  int t = blockIdx.x * blockDim.x + threadIdx.x;
  if (t < cnt) sel[idx[t]] = 1.f;
}

__global__ void k_fill(float* __restrict__ p, int count, float v) {
  int t = blockIdx.x * blockDim.x + threadIdx.x;
  if (t < count) p[t] = v;
}

// ---------------- launch ----------------
extern "C" void kernel_launch(void* const* d_in, const int* in_sizes, int n_in,
                              void* d_out, int out_size, void* d_ws, size_t ws_size,
                              hipStream_t stream) {
  const float* node_emb = (const float*)d_in[0];
  const float* X_ca     = (const float*)d_in[1];
  const float* gum      = (const float*)d_in[2];
  const float* Wa1_lr   = (const float*)d_in[3];
  const float* Wa2_lr   = (const float*)d_in[4];
  const float* Wv_lr    = (const float*)d_in[5];
  const float* Wo_lr    = (const float*)d_in[6];
  const float* Wg_lr    = (const float*)d_in[7];
  const float* Wa1_lo   = (const float*)d_in[8];
  const float* Wa2_lo   = (const float*)d_in[9];
  const float* Wv_lo    = (const float*)d_in[10];
  const float* Wo_lo    = (const float*)d_in[11];
  const float* Wg_lo    = (const float*)d_in[12];
  const unsigned char* xmask = (const unsigned char*)d_in[13];
  const int* noised     = (const int*)d_in[14];
  int n_noised          = in_sizes[14];
  float* dout = (float*)d_out;

  char* ws = (char*)d_ws;
  auto alloc = [&](size_t bytes) { char* p = ws; ws += (bytes + 255) & ~(size_t)255; return p; };
  int*   sinks1 = (int*)  alloc((size_t)E1 * 4);
  int*   sinks2 = (int*)  alloc((size_t)E2 * 4);
  float* db     = (float*)alloc((size_t)E1 * 4);
  float* logits = (float*)alloc((size_t)E1 * HH * 4);
  float* segmax = (float*)alloc((size_t)NTOT * HH * 4);
  float* segsum = (float*)alloc((size_t)NTOT * HH * 4);
  float* msgacc = (float*)alloc((size_t)NTOT * 256 * 4);
  float* outb   = (float*)alloc((size_t)NTOT * 128 * 4);
  float* newX   = (float*)alloc((size_t)NTOT * 3 * 4);
  float* sel    = (float*)alloc((size_t)NTOT * 4);

  // sel mask
  hipMemsetAsync(sel, 0, (size_t)NTOT * 4, stream);
  k_sel_scatter<<<(n_noised + 255) / 256, 256, 0, stream>>>(noised, sel, n_noised);

  // ---- pass 1 (long range) ----
  k_graph<0><<<NTOT, 256, 0, stream>>>(X_ca, xmask, gum, sinks1);
  k_dist<<<E1 / 256, 256, 0, stream>>>(E1, EPN1, sinks1, X_ca, db);
  k_fill<<<(NTOT * HH + 255) / 256, 256, 0, stream>>>(segmax, NTOT * HH, -3e38f);
  hipMemsetAsync(segsum, 0, (size_t)NTOT * HH * 4, stream);
  hipMemsetAsync(msgacc, 0, (size_t)NTOT * 256 * 4, stream);
  k_logits<<<E1 / 64, 128, 0, stream>>>(E1, EPN1, sinks1, node_emb, sel, X_ca, db,
                                        Wa1_lr, Wa2_lr, 0.1f, logits, segmax);
  k_psum<<<(E1 * HH) / 256, 256, 0, stream>>>(E1, sinks1, db, 0.1f, segmax, logits, segsum);
  k_value<<<E1 / 16, 128, 0, stream>>>(E1, EPN1, sinks1, node_emb, sel, X_ca, Wv_lr,
                                       logits, segsum, 0.1f, msgacc);
  k_outproj<<<(NTOT + 255) / 256, 256, 0, stream>>>(1, msgacc, Wo_lr, Wg_lr, outb);
  k_newx<<<(NTOT + 255) / 256, 256, 0, stream>>>(X_ca, sel, outb, newX);

  // ---- pass 2 (local kNN on updated coords) ----
  k_graph<1><<<NTOT, 256, 0, stream>>>(newX, xmask, gum, sinks2);
  k_dist<<<E2 / 256, 256, 0, stream>>>(E2, EPN2, sinks2, newX, db);
  k_fill<<<(NTOT * HH + 255) / 256, 256, 0, stream>>>(segmax, NTOT * HH, -3e38f);
  hipMemsetAsync(segsum, 0, (size_t)NTOT * HH * 4, stream);
  hipMemsetAsync(msgacc, 0, (size_t)NTOT * 256 * 4, stream);
  k_logits<<<E2 / 64, 128, 0, stream>>>(E2, EPN2, sinks2, node_emb, sel, newX, db,
                                        Wa1_lo, Wa2_lo, 1e-6f, logits, segmax);
  k_psum<<<(E2 * HH) / 256, 256, 0, stream>>>(E2, sinks2, db, 1e-6f, segmax, logits, segsum);
  k_value<<<E2 / 16, 128, 0, stream>>>(E2, EPN2, sinks2, node_emb, sel, newX, Wv_lo,
                                       logits, segsum, 1e-6f, msgacc);
  k_outproj<<<(NTOT + 255) / 256, 256, 0, stream>>>(32, msgacc, Wo_lo, Wg_lo, outb);

  // ---- assemble outputs: new_X (N,3) then new_emb (N,4,32) ----
  k_final<<<(NTOT + 255) / 256, 256, 0, stream>>>(newX, node_emb, sel, outb, dout);
}
